// Node2VecGraphClassifier_35588099015135
// MI455X (gfx1250) — matrix operand order repaired and checked by reading, not compiled, not run on hardware
//
#include <hip/hip_runtime.h>

typedef __attribute__((ext_vector_type(16))) __bf16 v16bf;
typedef __attribute__((ext_vector_type(8)))  float  v8f;

// ---------------------------------------------------------------------------
// Degree / normalization:  dinv[i] = (1 + indeg(i))^-1/2   (self loops incl.)
// ---------------------------------------------------------------------------
__global__ void deg_init_kernel(float* deg, int N) {
  int i = blockIdx.x * blockDim.x + threadIdx.x;
  if (i < N) deg[i] = 1.0f;  // self loop
}

__global__ void deg_count_kernel(const int* __restrict__ dst, float* deg, int E) {
  int i = blockIdx.x * blockDim.x + threadIdx.x;
  if (i < E) atomicAdd(&deg[dst[i]], 1.0f);
}

__global__ void deg_rsqrt_kernel(float* deg, int N) {
  int i = blockIdx.x * blockDim.x + threadIdx.x;
  if (i < N) deg[i] = rsqrtf(deg[i]);
}

// ---------------------------------------------------------------------------
// Pack fp32 weight matrix W[K,Nout] into bf16 WMMA B-fragment layout:
//   Wp[((tn*K32 + kk)*32 + lane)*16 + e] = bf16( W[k, n] )
//   n = tn*16 + (lane&15),  k = kk*32 + ((lane>>4)<<4) + e
// Each lane's 16-element fragment is then one contiguous 32B load.
// ---------------------------------------------------------------------------
__global__ void pack_w_kernel(const float* __restrict__ W, __bf16* __restrict__ Wp,
                              int K, int Nout) {
  int idx = blockIdx.x * blockDim.x + threadIdx.x;
  if (idx >= K * Nout) return;
  int e    = idx & 15;
  int l    = (idx >> 4) & 31;
  int rest = idx >> 9;             // tn*K32 + kk
  int K32  = K >> 5;
  int tn   = rest / K32;
  int kk   = rest - tn * K32;
  int n = tn * 16 + (l & 15);
  int k = kk * 32 + ((l >> 4) << 4) + e;
  Wp[idx] = (__bf16)W[k * Nout + n];
}

// ---------------------------------------------------------------------------
// bf16 WMMA GEMM:  out[M,Nout] = A[M,K] @ W[K,Nout]  (+bias, optional ReLU)
// One wave computes a 16x64 output strip (4 accumulators): the A fragment is
// loaded once per k-step and reused by 4 back-to-back WMMAs. B fragments come
// from the packed bf16 layout as contiguous 32B loads. Nout must be 256 here.
// ---------------------------------------------------------------------------
template <int K>
__global__ void wmma_gemm_kernel(const float* __restrict__ A,
                                 const __bf16* __restrict__ Wp,
                                 float* __restrict__ out,
                                 const float* __restrict__ bias,
                                 int M, int Nout, int relu) {
  constexpr int K32 = K / 32;
  const int wave    = blockIdx.x * (blockDim.x >> 5) + (threadIdx.x >> 5);
  const int ngroups = Nout >> 6;                 // 64-column groups
  const int total   = (M >> 4) * ngroups;
  if (wave >= total) return;                     // wave-uniform: EXEC stays all-ones

  const int tileM = wave / ngroups;
  const int ng    = wave - tileM * ngroups;
  const int lane  = threadIdx.x & 31;
  const int hf    = lane >> 4;                   // lane half (0 / 1)
  const int l16   = lane & 15;
  const int m     = tileM * 16 + l16;            // A row owned by this lane

  v8f acc0 = {}, acc1 = {}, acc2 = {}, acc3 = {};
  const float* arow = A + m * K;                 // fits in 32-bit index math
  const int tnb = ng * 4;

#pragma unroll
  for (int kk = 0; kk < K32; ++kk) {
    // ---- A fragment (lanes 0-15: K=0..15; lanes 16-31: K=16..31)
    const int ka = kk * 32 + hf * 16;
    const float4 a0 = *(const float4*)(arow + ka);
    const float4 a1 = *(const float4*)(arow + ka + 4);
    const float4 a2 = *(const float4*)(arow + ka + 8);
    const float4 a3 = *(const float4*)(arow + ka + 12);
    v16bf af;
    af[0]  = (__bf16)a0.x; af[1]  = (__bf16)a0.y; af[2]  = (__bf16)a0.z; af[3]  = (__bf16)a0.w;
    af[4]  = (__bf16)a1.x; af[5]  = (__bf16)a1.y; af[6]  = (__bf16)a1.z; af[7]  = (__bf16)a1.w;
    af[8]  = (__bf16)a2.x; af[9]  = (__bf16)a2.y; af[10] = (__bf16)a2.z; af[11] = (__bf16)a2.w;
    af[12] = (__bf16)a3.x; af[13] = (__bf16)a3.y; af[14] = (__bf16)a3.z; af[15] = (__bf16)a3.w;

    // ---- B fragments: contiguous 32B per lane from the packed layout
    const v16bf b0 = *(const v16bf*)(Wp + (((tnb + 0) * K32 + kk) * 32 + lane) * 16);
    const v16bf b1 = *(const v16bf*)(Wp + (((tnb + 1) * K32 + kk) * 32 + lane) * 16);
    const v16bf b2 = *(const v16bf*)(Wp + (((tnb + 2) * K32 + kk) * 32 + lane) * 16);
    const v16bf b3 = *(const v16bf*)(Wp + (((tnb + 3) * K32 + kk) * 32 + lane) * 16);

    acc0 = __builtin_amdgcn_wmma_f32_16x16x32_bf16(false, af, false, b0, (short)0, acc0, false, false);
    acc1 = __builtin_amdgcn_wmma_f32_16x16x32_bf16(false, af, false, b1, (short)0, acc1, false, false);
    acc2 = __builtin_amdgcn_wmma_f32_16x16x32_bf16(false, af, false, b2, (short)0, acc2, false, false);
    acc3 = __builtin_amdgcn_wmma_f32_16x16x32_bf16(false, af, false, b3, (short)0, acc3, false, false);
  }

  // ---- store 4 tiles (C/D layout: lane half selects M+8)
#pragma unroll
  for (int j = 0; j < 4; ++j) {
    const v8f acc = (j == 0) ? acc0 : (j == 1) ? acc1 : (j == 2) ? acc2 : acc3;
    const int n = (tnb + j) * 16 + l16;
    const float bn = bias ? bias[n] : 0.0f;
#pragma unroll
    for (int v = 0; v < 8; ++v) {
      const int row = tileM * 16 + v + hf * 8;
      float val = acc[v] + bn;
      if (relu) val = fmaxf(val, 0.0f);
      out[row * Nout + n] = val;
    }
  }
}

// ---------------------------------------------------------------------------
// Aggregation:  out = D^-1/2 (A+I) D^-1/2 h   (D = 256 channels, float4 wide)
// ---------------------------------------------------------------------------
__global__ void self_loop_kernel(const float* __restrict__ h,
                                 const float* __restrict__ dinv,
                                 float* __restrict__ out, int N) {
  int idx = blockIdx.x * blockDim.x + threadIdx.x;  // N*64 threads
  int node = idx >> 6;
  if (node >= N) return;
  int c = (idx & 63) << 2;
  float s = dinv[node]; s = s * s;
  float4 v = *(const float4*)(h + node * 256 + c);
  float4 o = make_float4(v.x * s, v.y * s, v.z * s, v.w * s);
  *(float4*)(out + node * 256 + c) = o;
}

__global__ void edge_scatter_kernel(const int* __restrict__ src,
                                    const int* __restrict__ dst,
                                    const float* __restrict__ dinv,
                                    const float* __restrict__ h,
                                    float* __restrict__ out, int E) {
  int tid = blockIdx.x * blockDim.x + threadIdx.x;  // E*64 threads
  int e = tid >> 6;
  if (e >= E) return;
  int c = (tid & 63) << 2;
  int s = src[e], d = dst[e];
  float norm = dinv[s] * dinv[d];
  float4 v = *(const float4*)(h + s * 256 + c);
  float* o = out + d * 256 + c;
  atomicAdd(o + 0, v.x * norm);
  atomicAdd(o + 1, v.y * norm);
  atomicAdd(o + 2, v.z * norm);
  atomicAdd(o + 3, v.w * norm);
}

__global__ void bias_relu_kernel(float* __restrict__ h,
                                 const float* __restrict__ b, int N) {
  int idx = blockIdx.x * blockDim.x + threadIdx.x;  // N*64 threads
  int node = idx >> 6;
  if (node >= N) return;
  int c = (idx & 63) << 2;
  float4 v = *(float4*)(h + node * 256 + c);
  float4 bb = *(const float4*)(b + c);
  v.x = fmaxf(v.x + bb.x, 0.0f);
  v.y = fmaxf(v.y + bb.y, 0.0f);
  v.z = fmaxf(v.z + bb.z, 0.0f);
  v.w = fmaxf(v.w + bb.w, 0.0f);
  *(float4*)(h + node * 256 + c) = v;
}

// ---------------------------------------------------------------------------
// Global mean pooling over batch ids (64 graphs x 256 channels)
// ---------------------------------------------------------------------------
__global__ void pool_zero_kernel(float* g, float* cnt) {
  int i = blockIdx.x * blockDim.x + threadIdx.x;
  if (i < 64 * 256) g[i] = 0.0f;
  if (i < 64) cnt[i] = 0.0f;
}

__global__ void pool_accum_kernel(const float* __restrict__ h,
                                  const int* __restrict__ batch,
                                  float* g, float* cnt, int N) {
  int idx = blockIdx.x * blockDim.x + threadIdx.x;  // N*64 threads
  int node = idx >> 6;
  if (node >= N) return;
  int c = (idx & 63) << 2;
  int b = batch[node];
  float4 v = *(const float4*)(h + node * 256 + c);
  float* go = g + b * 256 + c;
  atomicAdd(go + 0, v.x);
  atomicAdd(go + 1, v.y);
  atomicAdd(go + 2, v.z);
  atomicAdd(go + 3, v.w);
  if (c == 0) atomicAdd(&cnt[b], 1.0f);
}

__global__ void pool_div_kernel(float* g, const float* cnt) {
  int i = blockIdx.x * blockDim.x + threadIdx.x;
  if (i < 64 * 256) g[i] /= fmaxf(cnt[i >> 8], 1.0f);
}

// ---------------------------------------------------------------------------
// Final tiny FC: [64,256] @ [256,5] + b  -> d_out
// ---------------------------------------------------------------------------
__global__ void fc2_kernel(const float* __restrict__ g,
                           const float* __restrict__ W,
                           const float* __restrict__ b,
                           float* __restrict__ out) {
  int i = blockIdx.x * blockDim.x + threadIdx.x;
  if (i >= 64 * 5) return;
  int row = i / 5, col = i - row * 5;
  float s = b[col];
  const float* gr = g + row * 256;
#pragma unroll 8
  for (int k = 0; k < 256; ++k) s += gr[k] * W[k * 5 + col];
  out[i] = s;
}

static inline int cdiv(long long a, int b) { return (int)((a + b - 1) / b); }

extern "C" void kernel_launch(void* const* d_in, const int* in_sizes, int n_in,
                              void* d_out, int out_size, void* d_ws, size_t ws_size,
                              hipStream_t stream) {
  const float* x     = (const float*)d_in[0];
  const int*   ei    = (const int*)d_in[1];
  const int*   batch = (const int*)d_in[2];
  const float* W0 = (const float*)d_in[3];
  const float* b0 = (const float*)d_in[4];
  const float* W1 = (const float*)d_in[5];
  const float* b1 = (const float*)d_in[6];
  const float* W2 = (const float*)d_in[7];
  const float* b2 = (const float*)d_in[8];
  const float* Wc1 = (const float*)d_in[9];
  const float* bc1 = (const float*)d_in[10];
  const float* Wc2 = (const float*)d_in[11];
  const float* bc2 = (const float*)d_in[12];

  const int N = in_sizes[2];       // 50000 (multiple of 16)
  const int E = in_sizes[1] / 2;   // 500000
  const int D = 256;
  const int* src = ei;
  const int* dst = ei + E;

  float* ws   = (float*)d_ws;
  float* dinv = ws;                           // N
  float* bufA = dinv + N;                     // N*256  (GEMM output)
  float* bufB = bufA + (size_t)N * D;         // N*256  (aggregated / activations)
  float* g    = bufB + (size_t)N * D;         // 64*256
  float* cnt  = g + 64 * D;                   // 64
  float* g2   = cnt + 64;                     // 64*256

  // packed bf16 weights (aligned to 64B)
  uintptr_t p = (uintptr_t)(g2 + 64 * D);
  p = (p + 63) & ~(uintptr_t)63;
  __bf16* pw0  = (__bf16*)p;                  // 128*256
  __bf16* pw1  = pw0 + 128 * 256;             // 256*256
  __bf16* pw2  = pw1 + 256 * 256;             // 256*256
  __bf16* pwc1 = pw2 + 256 * 256;             // 256*256

  // ---- pack weights into WMMA B-fragment layout (bf16)
  pack_w_kernel<<<cdiv(128 * 256, 256), 256, 0, stream>>>(W0,  pw0,  128, D);
  pack_w_kernel<<<cdiv(256 * 256, 256), 256, 0, stream>>>(W1,  pw1,  256, D);
  pack_w_kernel<<<cdiv(256 * 256, 256), 256, 0, stream>>>(W2,  pw2,  256, D);
  pack_w_kernel<<<cdiv(256 * 256, 256), 256, 0, stream>>>(Wc1, pwc1, 256, D);

  // ---- degree / normalization (shared by all 3 layers)
  deg_init_kernel<<<cdiv(N, 256), 256, 0, stream>>>(dinv, N);
  deg_count_kernel<<<cdiv(E, 256), 256, 0, stream>>>(dst, dinv, E);
  deg_rsqrt_kernel<<<cdiv(N, 256), 256, 0, stream>>>(dinv, N);

  const int wavesL = (N / 16) * (D / 64);    // 12500 waves (16x64 strip each)
  const int wavesC = (64 / 16) * (D / 64);   // 16 waves

  // ---- layer 0: x[N,128] @ W0 -> bufA ; aggregate -> bufB ; +b0, ReLU
  wmma_gemm_kernel<128><<<cdiv(wavesL, 8), 256, 0, stream>>>(x, pw0, bufA, nullptr, N, D, 0);
  self_loop_kernel<<<cdiv((long long)N * 64, 256), 256, 0, stream>>>(bufA, dinv, bufB, N);
  edge_scatter_kernel<<<cdiv((long long)E * 64, 256), 256, 0, stream>>>(src, dst, dinv, bufA, bufB, E);
  bias_relu_kernel<<<cdiv((long long)N * 64, 256), 256, 0, stream>>>(bufB, b0, N);

  // ---- layer 1
  wmma_gemm_kernel<256><<<cdiv(wavesL, 8), 256, 0, stream>>>(bufB, pw1, bufA, nullptr, N, D, 0);
  self_loop_kernel<<<cdiv((long long)N * 64, 256), 256, 0, stream>>>(bufA, dinv, bufB, N);
  edge_scatter_kernel<<<cdiv((long long)E * 64, 256), 256, 0, stream>>>(src, dst, dinv, bufA, bufB, E);
  bias_relu_kernel<<<cdiv((long long)N * 64, 256), 256, 0, stream>>>(bufB, b1, N);

  // ---- layer 2
  wmma_gemm_kernel<256><<<cdiv(wavesL, 8), 256, 0, stream>>>(bufB, pw2, bufA, nullptr, N, D, 0);
  self_loop_kernel<<<cdiv((long long)N * 64, 256), 256, 0, stream>>>(bufA, dinv, bufB, N);
  edge_scatter_kernel<<<cdiv((long long)E * 64, 256), 256, 0, stream>>>(src, dst, dinv, bufA, bufB, E);
  bias_relu_kernel<<<cdiv((long long)N * 64, 256), 256, 0, stream>>>(bufB, b2, N);

  // ---- global mean pool
  pool_zero_kernel<<<cdiv(64 * 256, 256), 256, 0, stream>>>(g, cnt);
  pool_accum_kernel<<<cdiv((long long)N * 64, 256), 256, 0, stream>>>(bufB, batch, g, cnt, N);
  pool_div_kernel<<<cdiv(64 * 256, 256), 256, 0, stream>>>(g, cnt);

  // ---- classifier: relu(g @ Wc1 + bc1) @ Wc2 + bc2
  wmma_gemm_kernel<256><<<cdiv(wavesC, 8), 256, 0, stream>>>(g, pwc1, g2, bc1, 64, D, 1);
  fc2_kernel<<<2, 256, 0, stream>>>(g2, Wc2, bc2, (float*)d_out);
}
